// SimpleAttention_86157043957990
// MI455X (gfx1250) — compile-verified, matrix operand order checked by reading, not run on hardware
//
#include <hip/hip_runtime.h>
#include <hip/hip_bf16.h>

// ---------- types ----------
typedef __bf16 v16bf __attribute__((ext_vector_type(16)));
typedef float  v8f   __attribute__((ext_vector_type(8)));
typedef unsigned int u32x4 __attribute__((ext_vector_type(4)));
typedef float  f32x4 __attribute__((ext_vector_type(4)));

union BFrag {
    v16bf v;
    u32x4 u[2];
};

__device__ __forceinline__ __bf16 f2bf(float f) {
    union { float f; unsigned int u; } cv;
    cv.f = f;
    unsigned int r = cv.u + 0x7FFFu + ((cv.u >> 16) & 1u);  // round-to-nearest-even
    unsigned short h = (unsigned short)(r >> 16);
    __bf16 out;
    __builtin_memcpy(&out, &h, 2);
    return out;
}

// Async copy 16B global -> LDS (CDNA5 GLOBAL_LOAD_ASYNC_TO_LDS_B128, ASYNCcnt).
__device__ __forceinline__ void async_copy_b128(unsigned lds_off, const void* gptr) {
    asm volatile("global_load_async_to_lds_b128 %0, %1, off"
                 :: "v"(lds_off), "v"(gptr) : "memory");
}
__device__ __forceinline__ void wait_async() {
    asm volatile("s_wait_asynccnt 0" ::: "memory");
}
__device__ __forceinline__ unsigned lds_addr(const void* p) {
    return (unsigned)(unsigned long long)p;
}

#define N_ROWS 8192
#define D_FEAT 1024

// ---------------------------------------------------------------------------
// Kernel 1: row L2-norm; emit bf16 normalized feats [N,D] and bf16 V^T [D,N].
// ---------------------------------------------------------------------------
__global__ void __launch_bounds__(256, 1)
k_norm_convert(const float* __restrict__ X,
               __bf16* __restrict__ Nbf,   // [N, D]
               __bf16* __restrict__ Vt) {  // [D, N]
    const int row = blockIdx.x;
    const float* x = X + (size_t)row * D_FEAT;

    float ss = 0.f;
    for (int i = threadIdx.x; i < D_FEAT; i += 256) {
        float v = x[i];
        ss += v * v;
    }
    __shared__ float buf[256];
    buf[threadIdx.x] = ss;
    __syncthreads();
    for (int t = 128; t > 0; t >>= 1) {
        if (threadIdx.x < t) buf[threadIdx.x] += buf[threadIdx.x + t];
        __syncthreads();
    }
    float inv = 1.0f / fmaxf(sqrtf(buf[0]), 1e-12f);

    for (int i = threadIdx.x; i < D_FEAT; i += 256) {
        float v = x[i];
        Nbf[(size_t)row * D_FEAT + i] = f2bf(v * inv);
        Vt[(size_t)i * N_ROWS + row] = f2bf(v);
    }
}

// ---------------------------------------------------------------------------
// Kernel 2: S = Nf @ Nf^T, diag = -inf.
// Block tile 128x128, 8 waves (2x4), wave tile 64x32 (4x2 frags, 8 WMMA/K-step).
// A/B tiles double-buffered in LDS via async copies.  K-loop is kept
// single-body (no unroll) so WMMA accumulators stay pinned in place.
// ---------------------------------------------------------------------------
#define G1_BM 128
#define G1_BN 128
#define G1_BK 32
#define G1_LDK 40   // padded row stride (elements): 80B, spreads LDS banks

__global__ void __launch_bounds__(256, 1)
k_gemm_nnt(const __bf16* __restrict__ A,  // [N, D]
           float* __restrict__ S) {       // [N, N]
    __shared__ alignas(16) __bf16 Ash[2][G1_BM][G1_LDK];
    __shared__ alignas(16) __bf16 Bsh[2][G1_BN][G1_LDK];

    const int tid   = threadIdx.x;
    const int widx  = tid >> 5;
    const int lane  = tid & 31;
    const int hi    = lane >> 4;
    const int l15   = lane & 15;
    const int waveM = widx >> 2;   // 0..1
    const int waveN = widx & 3;    // 0..3

    const int bm = (blockIdx.x / (N_ROWS / G1_BN)) * G1_BM;
    const int bn = (blockIdx.x % (N_ROWS / G1_BN)) * G1_BN;

    // Stage both 128x32 bf16 tiles: 512 chunks of 16B each, 2 per thread.
    auto stage = [&](int bufi, int k0) {
#pragma unroll
        for (int t = 0; t < 2; ++t) {
            int c   = tid + t * 256;      // 0..511
            int row = c >> 2;
            int col = (c & 3) * 8;
            async_copy_b128(lds_addr(&Ash[bufi][row][col]),
                            A + (size_t)(bm + row) * D_FEAT + k0 + col);
            async_copy_b128(lds_addr(&Bsh[bufi][row][col]),
                            A + (size_t)(bn + row) * D_FEAT + k0 + col);
        }
    };

    const int nK = D_FEAT / G1_BK;  // 32
    stage(0, 0);
    wait_async();
    __syncthreads();

    // Per-wave fragment base pointers into buffer 0; buffer select at runtime.
    const __bf16* aBase = &Ash[0][waveM * 64 + l15][hi * 8];
    const __bf16* bBase = &Bsh[0][waveN * 32 + l15][hi * 8];
    const int aBufStride = G1_BM * G1_LDK;   // elements per buffer
    const int bBufStride = G1_BN * G1_LDK;

    v8f c[4][2] = {};
#pragma clang loop unroll(disable)
    for (int kk = 0; kk < nK; ++kk) {
        const int cur = kk & 1;
        if (kk + 1 < nK) stage(cur ^ 1, (kk + 1) * G1_BK);

        const __bf16* aP = aBase + cur * aBufStride;
        const __bf16* bP = bBase + cur * bBufStride;

        BFrag bf[2];
#pragma unroll
        for (int fn = 0; fn < 2; ++fn) {
            const __bf16* p = bP + fn * 16 * G1_LDK;
            bf[fn].u[0] = *(const u32x4*)(p);
            bf[fn].u[1] = *(const u32x4*)(p + 16);
        }
#pragma unroll
        for (int fm = 0; fm < 4; ++fm) {
            BFrag af;
            const __bf16* p = aP + fm * 16 * G1_LDK;
            af.u[0] = *(const u32x4*)(p);
            af.u[1] = *(const u32x4*)(p + 16);
#pragma unroll
            for (int fn = 0; fn < 2; ++fn) {
                c[fm][fn] = __builtin_amdgcn_wmma_f32_16x16x32_bf16(
                    false, af.v, false, bf[fn].v, (short)0, c[fm][fn], false, false);
            }
        }
        wait_async();      // our async fills for the next buffer are done
        __syncthreads();   // everyone finished reading 'cur' / filling 'nxt'
    }

    // C/D layout: VGPR r: lanes 0-15 -> (M=r, N=lane); lanes 16-31 -> (M=r+8).
#pragma unroll
    for (int fm = 0; fm < 4; ++fm) {
        const int gm0 = bm + waveM * 64 + fm * 16 + hi * 8;
#pragma unroll
        for (int fn = 0; fn < 2; ++fn) {
            const int gn = bn + waveN * 32 + fn * 16 + l15;
#pragma unroll
            for (int r = 0; r < 8; ++r) {
                float v = c[fm][fn][r];
                if (gm0 + r == gn) v = -__builtin_inff();   // self-similarity mask
                S[(size_t)(gm0 + r) * N_ROWS + gn] = v;
            }
        }
    }
}

// ---------------------------------------------------------------------------
// Kernel 3: in-place row softmax over S [N, N].
// ---------------------------------------------------------------------------
__global__ void __launch_bounds__(256, 1)
k_softmax_rows(float* __restrict__ S) {
    const int row = blockIdx.x;
    float* s = S + (size_t)row * N_ROWS;
    __shared__ float buf[256];

    float m = -__builtin_inff();
    for (int i = threadIdx.x; i < N_ROWS; i += 256) m = fmaxf(m, s[i]);
    buf[threadIdx.x] = m;
    __syncthreads();
    for (int t = 128; t > 0; t >>= 1) {
        if (threadIdx.x < t) buf[threadIdx.x] = fmaxf(buf[threadIdx.x], buf[threadIdx.x + t]);
        __syncthreads();
    }
    m = buf[0];
    __syncthreads();

    float sum = 0.f;
    for (int i = threadIdx.x; i < N_ROWS; i += 256) sum += __expf(s[i] - m);
    buf[threadIdx.x] = sum;
    __syncthreads();
    for (int t = 128; t > 0; t >>= 1) {
        if (threadIdx.x < t) buf[threadIdx.x] += buf[threadIdx.x + t];
        __syncthreads();
    }
    const float inv = 1.0f / buf[0];

    for (int i = threadIdx.x; i < N_ROWS; i += 256) s[i] = __expf(s[i] - m) * inv;
}

// ---------------------------------------------------------------------------
// Kernel 4: O = P @ V.  Block = 32 rows x ALL 1024 cols so the 256MB attn
// matrix is streamed exactly once (it does not fit in the 192MB L2).
// P tile (32x32 f32) async-staged to LDS, converted to bf16 at frag read.
// 8 waves each own a 128-col slab: 2(M) x 8(N) frags, 16 WMMA/K-step.
// Vt (16MB) stays L2-resident and is read directly from global.
// ---------------------------------------------------------------------------
#define G2_LDK 36   // padded f32 row stride (144B)

__global__ void __launch_bounds__(256, 1)
k_gemm_av(const float* __restrict__ P,    // [N, N] softmaxed
          const __bf16* __restrict__ Vt,  // [D, N]
          float* __restrict__ O) {        // [N, D]
    __shared__ alignas(16) float Psh[2][32][G2_LDK];

    const int tid  = threadIdx.x;
    const int widx = tid >> 5;
    const int lane = tid & 31;
    const int hi   = lane >> 4;
    const int l15  = lane & 15;

    const int rowBase = blockIdx.x * 32;
    const int colBase = widx * 128;

    // Stage 32x32 f32 tile: 256 chunks of 16B, exactly one per thread.
    auto stageA = [&](int bufi, int k0) {
        int row = tid >> 3;        // 0..31
        int col = (tid & 7) * 4;   // 0..28 (floats)
        async_copy_b128(lds_addr(&Psh[bufi][row][col]),
                        P + (size_t)(rowBase + row) * N_ROWS + k0 + col);
    };

    const __bf16* vtb = Vt + (size_t)(colBase + l15) * N_ROWS + hi * 8;
    const float* pBase = &Psh[0][l15][hi * 8];
    const int pBufStride = 32 * G2_LDK;   // f32 elements per buffer

    const int nK = N_ROWS / 32;   // 256
    stageA(0, 0);
    wait_async();
    __syncthreads();

    v8f c[2][8] = {};
#pragma clang loop unroll(disable)
    for (int kk = 0; kk < nK; ++kk) {
        const int cur = kk & 1;
        if (kk + 1 < nK) stageA(cur ^ 1, (kk + 1) * 32);

        // A fragments: read f32 from LDS, convert to bf16 in registers.
        BFrag af[2];
#pragma unroll
        for (int fm = 0; fm < 2; ++fm) {
            const float* p = pBase + cur * pBufStride + fm * 16 * G2_LDK;
            f32x4 a0 = *(const f32x4*)(p);
            f32x4 a1 = *(const f32x4*)(p + 4);
            f32x4 a2 = *(const f32x4*)(p + 16);
            f32x4 a3 = *(const f32x4*)(p + 20);
#pragma unroll
            for (int t = 0; t < 4; ++t) {
                af[fm].v[t]      = f2bf(a0[t]);
                af[fm].v[4 + t]  = f2bf(a1[t]);
                af[fm].v[8 + t]  = f2bf(a2[t]);
                af[fm].v[12 + t] = f2bf(a3[t]);
            }
        }
#pragma unroll
        for (int fn = 0; fn < 8; ++fn) {
            const __bf16* bp = vtb + (size_t)fn * 16 * N_ROWS + kk * 32;
            BFrag b;
            b.u[0] = *(const u32x4*)(bp);
            b.u[1] = *(const u32x4*)(bp + 16);
            c[0][fn] = __builtin_amdgcn_wmma_f32_16x16x32_bf16(
                false, af[0].v, false, b.v, (short)0, c[0][fn], false, false);
            c[1][fn] = __builtin_amdgcn_wmma_f32_16x16x32_bf16(
                false, af[1].v, false, b.v, (short)0, c[1][fn], false, false);
        }
        wait_async();
        __syncthreads();
    }

#pragma unroll
    for (int fm = 0; fm < 2; ++fm) {
        const int gm0 = rowBase + fm * 16 + hi * 8;
#pragma unroll
        for (int fn = 0; fn < 8; ++fn) {
            const int gn = colBase + fn * 16 + l15;
#pragma unroll
            for (int r = 0; r < 8; ++r) {
                O[(size_t)(gm0 + r) * D_FEAT + gn] = c[fm][fn][r];
            }
        }
    }
}

// ---------------------------------------------------------------------------
// d_out layout: [out (8192*1024 f32)] [attn (8192*8192 f32)]
// d_ws  layout: [Nbf bf16 8192*1024 = 16MB] [Vt bf16 1024*8192 = 16MB]
// ---------------------------------------------------------------------------
extern "C" void kernel_launch(void* const* d_in, const int* in_sizes, int n_in,
                              void* d_out, int out_size, void* d_ws, size_t ws_size,
                              hipStream_t stream) {
    const float* img = (const float*)d_in[0];
    float* out  = (float*)d_out;                              // [8192, 1024]
    float* attn = (float*)d_out + (size_t)N_ROWS * D_FEAT;    // [8192, 8192]

    __bf16* Nbf = (__bf16*)d_ws;                                                     // [8192,1024]
    __bf16* Vt  = (__bf16*)((char*)d_ws + (size_t)N_ROWS * D_FEAT * sizeof(__bf16)); // [1024,8192]

    // 1) normalize + bf16 conversions
    k_norm_convert<<<N_ROWS, 256, 0, stream>>>(img, Nbf, Vt);

    // 2) sims = Nf @ Nf^T with -inf diagonal: (8192/128)^2 = 4096 blocks
    k_gemm_nnt<<<(N_ROWS / G1_BM) * (N_ROWS / G1_BN), 256, 0, stream>>>(Nbf, attn);

    // 3) row softmax in place
    k_softmax_rows<<<N_ROWS, 256, 0, stream>>>(attn);

    // 4) out = attn @ V: 8192/32 = 256 blocks, each covers 32 rows x 1024 cols
    k_gemm_av<<<N_ROWS / 32, 256, 0, stream>>>(attn, Vt, out);
}
